// SAM_3882650436600
// MI455X (gfx1250) — matrix operand order repaired and checked by reading, not compiled, not run on hardware
//
#include <hip/hip_runtime.h>
#include <hip/hip_bf16.h>

// ---------------------------------------------------------------------------
// SAM attention-mask pipeline on gfx1250 (MI455X):
//   conv3x3(32->32) -> IRNN4 -> conv3x3(128->32) -> IRNN4 -> conv3x3+relu
//   -> conv1x1 + sigmoid
// Convs are implicit GEMMs on v_wmma_f32_16x16x32_bf16 (f32 accumulate).
// Intermediates are NHWC bf16. Halo tiles are staged to LDS with
// global_load_async_to_lds_b128 (ASYNCcnt). Each wave computes two 16-pixel
// strips at once so every B fragment feeds two WMMAs.
// ---------------------------------------------------------------------------

typedef __attribute__((ext_vector_type(16))) __bf16 bf16x16;
typedef __attribute__((ext_vector_type(8)))  __bf16 bf16x8;
typedef __attribute__((ext_vector_type(8)))  float  f32x8;

#define BB   8
#define HH   256
#define WWD  256
#define COUT 32

#if __has_builtin(__builtin_amdgcn_global_load_async_to_lds_b128)
#define HAVE_ASYNC_LDS 1
#else
#define HAVE_ASYNC_LDS 0
#endif

#if HAVE_ASYNC_LDS
// Builtin signature (from clang diagnostics): first arg is a pointer to a
// 16-byte int vector in addrspace(1); LDS side is the same vector type in
// addrspace(3).
typedef __attribute__((__vector_size__(4 * sizeof(int)))) int i32x4v;
typedef __attribute__((address_space(1))) i32x4v* gptr_b128;
typedef __attribute__((address_space(3))) i32x4v* lptr_b128;
__device__ __forceinline__ gptr_b128 as_g128(const void* p) {
    return (gptr_b128)(uintptr_t)p;
}
__device__ __forceinline__ lptr_b128 as_l128(void* p) {
    // low 32 bits of a generic shared pointer are the LDS byte address
    return (lptr_b128)(uintptr_t)p;
}
#endif

__device__ __forceinline__ unsigned short f2bf(float f) {
    union { float f; unsigned u; } v; v.f = f;
    unsigned u = v.u;
    u += 0x7FFFu + ((u >> 16) & 1u);   // round to nearest even
    return (unsigned short)(u >> 16);
}
__device__ __forceinline__ float bf2f(unsigned short h) {
    union { unsigned u; float f; } v; v.u = ((unsigned)h) << 16;
    return v.f;
}

// --------------------------- weight fragment prep ---------------------------
// Destination layout: f = (((tap*KB + kb)*2 + nb)*32 + lane)*16 + j
//   B-matrix (K x N) per-lane layout: n = lane&15, k = (j&7) + 8*(lane>>4) + 16*(j>>3)
__global__ void k_prep_wfrag(const float* __restrict__ w,
                             unsigned short* __restrict__ wf,
                             int cin, int nfrag) {
    int f = blockIdx.x * blockDim.x + threadIdx.x;
    if (f >= nfrag) return;
    int j     = f & 15;
    int lane  = (f >> 4) & 31;
    int nb    = (f >> 9) & 1;
    int kbtap = f >> 10;
    int KB    = cin >> 5;
    int kb    = kbtap % KB;
    int tap   = kbtap / KB;
    int co = nb * 16 + (lane & 15);
    int k  = (j & 7) + ((lane >> 4) << 3) + ((j >> 3) << 4);
    int ci = kb * 32 + k;
    int kh = tap / 3, kw = tap % 3;
    wf[f] = f2bf(w[((co * cin + ci) * 3 + kh) * 3 + kw]);
}

// --------------------------- NCHW f32 -> NHWC bf16 --------------------------
__global__ void k_nchw_to_nhwc(const float* __restrict__ x,
                               unsigned short* __restrict__ xh) {
    long idx4   = (long)blockIdx.x * blockDim.x + threadIdx.x;
    long total4 = (long)BB * 32 * HH * WWD / 4;
    if (idx4 >= total4) return;
    const float4* x4 = (const float4*)x;
    float4 v = x4[idx4];
    long e = idx4 * 4;          // linear (b,c,h,w), w fastest
    int w = (int)(e % WWD);
    long t = e / WWD;
    int h = (int)(t % HH); t /= HH;
    int c = (int)(t % 32);
    int b = (int)(t / 32);
    long pbase = ((long)b * HH + h) * WWD;
    xh[(pbase + w + 0) * 32 + c] = f2bf(v.x);
    xh[(pbase + w + 1) * 32 + c] = f2bf(v.y);
    xh[(pbase + w + 2) * 32 + c] = f2bf(v.z);
    xh[(pbase + w + 3) * 32 + c] = f2bf(v.w);
}

// --------------------------- WMMA 3x3 conv ----------------------------------
// Tile: 8 rows x 16 cols of output per block (128 threads = 4 waves).
// LDS halo tile (10 x 18 x CIN) bf16, channel-last.
// Each wave computes rows wv and wv+4 simultaneously: 2 A fragments and
// 2 B fragments feed 4 WMMAs per (tap, kblock).
template<int CIN>
__global__ __launch_bounds__(128)
void k_conv3x3(const unsigned short* __restrict__ in,
               const unsigned short* __restrict__ wf,
               unsigned short* __restrict__ out,
               int do_relu) {
    constexpr int TH = 8, TW = 16;
    constexpr int KB = CIN / 32;
    constexpr int HL = TH + 2, WL = TW + 2;     // 10 x 18 halo
    __shared__ unsigned short tile[HL * WL * CIN];

    const int w0  = blockIdx.x * TW;
    const int h0  = blockIdx.y * TH;
    const int b   = blockIdx.z;
    const int tid = threadIdx.x;

    // Hint the memory system about this block's working set (global_prefetch_b8).
    __builtin_prefetch(in + (((long)b * HH + h0) * WWD + w0) * CIN, 0, 3);

    // ---- stage halo tile into LDS (zero padding outside image) ----
    constexpr int C8  = CIN / 8;                // 16B chunks along channels
    constexpr int NCH = HL * WL * C8;
    for (int c = tid; c < NCH; c += 128) {
        int cb = c % C8;
        int t  = c / C8;
        int ww = t % WL;
        int hh = t / WL;
        int gh = h0 - 1 + hh;
        int gw = w0 - 1 + ww;
        unsigned short* lp = tile + (hh * WL + ww) * CIN + cb * 8;
        if (gh >= 0 && gh < HH && gw >= 0 && gw < WWD) {
            const unsigned short* gp =
                in + (((long)b * HH + gh) * WWD + gw) * CIN + cb * 8;
#if HAVE_ASYNC_LDS
            __builtin_amdgcn_global_load_async_to_lds_b128(
                as_g128(gp), as_l128(lp), 0, 0);
#else
            *(uint4*)lp = *(const uint4*)gp;
#endif
        } else {
            *(uint4*)lp = make_uint4(0u, 0u, 0u, 0u);
        }
    }
#if HAVE_ASYNC_LDS
#if __has_builtin(__builtin_amdgcn_s_wait_asynccnt)
    __builtin_amdgcn_s_wait_asynccnt(0);
#else
    asm volatile("s_wait_asynccnt 0" ::: "memory");
#endif
#endif
    __syncthreads();

    const int lane = tid & 31;
    const int wv   = tid >> 5;
    const int m    = lane & 15;
    const int hi   = (lane & 16) ? 8 : 0;
    const int r0   = wv;
    const int r1   = wv + 4;

    f32x8 acc00 = {};   // row set r0, co 0-15
    f32x8 acc01 = {};   // row set r0, co 16-31
    f32x8 acc10 = {};   // row set r1, co 0-15
    f32x8 acc11 = {};   // row set r1, co 16-31
    #pragma unroll
    for (int tap = 0; tap < 9; ++tap) {
        const int kh = tap / 3, kw = tap % 3;
        #pragma unroll
        for (int kb = 0; kb < KB; ++kb) {
            const unsigned short* a0base =
                tile + ((r0 + kh) * WL + (m + kw)) * CIN + kb * 32;
            const unsigned short* a1base =
                tile + ((r1 + kh) * WL + (m + kw)) * CIN + kb * 32;
            bf16x8 p0 = *(const bf16x8*)(a0base + hi);
            bf16x8 p1 = *(const bf16x8*)(a0base + 16 + hi);
            bf16x8 q0 = *(const bf16x8*)(a1base + hi);
            bf16x8 q1 = *(const bf16x8*)(a1base + 16 + hi);
            bf16x16 a0, a1;
            #pragma unroll
            for (int i = 0; i < 8; ++i) {
                a0[i] = p0[i]; a0[8 + i] = p1[i];
                a1[i] = q0[i]; a1[8 + i] = q1[i];
            }
            const unsigned short* wb =
                wf + ((tap * KB + kb) * 2 * 32 + lane) * 16;
            bf16x16 b0 = *(const bf16x16*)(wb);
            bf16x16 b1 = *(const bf16x16*)(wb + 512);
            acc00 = __builtin_amdgcn_wmma_f32_16x16x32_bf16(
                        false, a0, false, b0, (short)0, acc00, false, false);
            acc01 = __builtin_amdgcn_wmma_f32_16x16x32_bf16(
                        false, a0, false, b1, (short)0, acc01, false, false);
            acc10 = __builtin_amdgcn_wmma_f32_16x16x32_bf16(
                        false, a1, false, b0, (short)0, acc10, false, false);
            acc11 = __builtin_amdgcn_wmma_f32_16x16x32_bf16(
                        false, a1, false, b1, (short)0, acc11, false, false);
        }
    }

    // ---- store D: lane holds co = nb*16 + (lane&15); rows rr + 8*(lane>=16)
    const int mofs = (lane & 16) ? 8 : 0;
    const int co   = lane & 15;
    #pragma unroll
    for (int rr = 0; rr < 8; ++rr) {
        int gw = w0 + rr + mofs;
        float v00 = acc00[rr], v01 = acc01[rr];
        float v10 = acc10[rr], v11 = acc11[rr];
        if (do_relu) {
            v00 = fmaxf(v00, 0.f); v01 = fmaxf(v01, 0.f);
            v10 = fmaxf(v10, 0.f); v11 = fmaxf(v11, 0.f);
        }
        long p0 = (((long)b * HH + (h0 + r0)) * WWD + gw) * COUT;
        long p1 = (((long)b * HH + (h0 + r1)) * WWD + gw) * COUT;
        out[p0 + co]      = f2bf(v00);
        out[p0 + 16 + co] = f2bf(v01);
        out[p1 + co]      = f2bf(v10);
        out[p1 + 16 + co] = f2bf(v11);
    }
}

// --------------------------- fused 4-direction IRNN -------------------------
// h_t = relu(x_t + w_c * h_{t-1} + b_c); 262144 independent lanes, 256 steps.
// dir is the slowest index so control flow is wave-uniform.
// dirs: 0=up(axisH,rev) 1=right(axisW,fwd) 2=down(axisH,fwd) 3=left(axisW,rev)
__global__ void k_irnn4(const unsigned short* __restrict__ y,
                        const float* __restrict__ wv4,
                        const float* __restrict__ bv4,
                        unsigned short* __restrict__ z) {
    int tid = blockIdx.x * blockDim.x + threadIdx.x;
    int c   = tid & 31;
    int q   = tid >> 5;
    int fix = q & 255;          // fixed h (axis_w) or fixed w (axis_h)
    int b   = (q >> 8) & 7;
    int dir = q >> 11;          // 0..3
    int axis_w = (dir == 1) | (dir == 3);
    int rev    = (dir == 0) | (dir == 3);
    float wc = wv4[dir * 32 + c];
    float bc = bv4[dir * 32 + c];
    long ybase, ystride, zbase, zstride;
    if (axis_w) {
        ybase   = (((long)b * HH + fix) * WWD) * 32 + c;
        ystride = 32;
        zbase   = (((long)b * HH + fix) * WWD) * 128 + dir * 32 + c;
        zstride = 128;
    } else {
        ybase   = (((long)b * HH) * WWD + fix) * 32 + c;
        ystride = (long)WWD * 32;
        zbase   = (((long)b * HH) * WWD + fix) * 128 + dir * 32 + c;
        zstride = (long)WWD * 128;
    }
    float hstate = 0.f;
    for (int t = 0; t < 256; ++t) {
        int pos = rev ? (255 - t) : t;
        float x = bf2f(y[ybase + (long)pos * ystride]);
        hstate = fmaxf(x + wc * hstate + bc, 0.f);
        z[zbase + (long)pos * zstride] = f2bf(hstate);
    }
}

// --------------------------- 1x1 conv + sigmoid -----------------------------
__global__ void k_out(const unsigned short* __restrict__ y,
                      const float* __restrict__ wout,
                      float* __restrict__ outm) {
    long p = (long)blockIdx.x * blockDim.x + threadIdx.x;
    if (p >= (long)BB * HH * WWD) return;
    const unsigned short* yp = y + p * 32;
    float s = 0.f;
    #pragma unroll
    for (int c = 0; c < 32; ++c) s += bf2f(yp[c]) * wout[c];
    outm[p] = 1.f / (1.f + expf(-s));
}

// ---------------------------------------------------------------------------
extern "C" void kernel_launch(void* const* d_in, const int* in_sizes, int n_in,
                              void* d_out, int out_size, void* d_ws, size_t ws_size,
                              hipStream_t stream) {
    const float* x    = (const float*)d_in[0];
    const float* w_in = (const float*)d_in[1];
    const float* w_c2 = (const float*)d_in[2];
    const float* w_c3 = (const float*)d_in[3];
    const float* w_o  = (const float*)d_in[4];
    const float* i1w  = (const float*)d_in[5];
    const float* i1b  = (const float*)d_in[6];
    const float* i2w  = (const float*)d_in[7];
    const float* i2b  = (const float*)d_in[8];
    float* mask = (float*)d_out;

    char* ws = (char*)d_ws;
    size_t off = 0;
    auto alloc = [&](size_t bytes) {
        char* p = ws + off;
        off = (off + bytes + 255) & ~(size_t)255;
        return p;
    };
    unsigned short* xh  = (unsigned short*)alloc((size_t)BB * HH * WWD * 32 * 2);
    unsigned short* y   = (unsigned short*)alloc((size_t)BB * HH * WWD * 32 * 2);
    unsigned short* z   = (unsigned short*)alloc((size_t)BB * HH * WWD * 128 * 2);
    unsigned short* w1f = (unsigned short*)alloc((size_t)9 * 32 * 32 * 2);
    unsigned short* w2f = (unsigned short*)alloc((size_t)9 * 128 * 32 * 2);
    unsigned short* w3f = (unsigned short*)alloc((size_t)9 * 128 * 32 * 2);

    // weight fragment prep (bf16, WMMA B layout)
    k_prep_wfrag<<<(9216 + 255) / 256, 256, 0, stream>>>(w_in, w1f, 32, 9216);
    k_prep_wfrag<<<(36864 + 255) / 256, 256, 0, stream>>>(w_c2, w2f, 128, 36864);
    k_prep_wfrag<<<(36864 + 255) / 256, 256, 0, stream>>>(w_c3, w3f, 128, 36864);

    // input layout conversion
    long tot4 = (long)BB * 32 * HH * WWD / 4;
    k_nchw_to_nhwc<<<(unsigned)((tot4 + 255) / 256), 256, 0, stream>>>(x, xh);

    dim3 cgrid(WWD / 16, HH / 8, BB);

    // conv_in
    k_conv3x3<32><<<cgrid, 128, 0, stream>>>(xh, w1f, y, 0);

    // irnn1 (all 4 directions fused)
    k_irnn4<<<262144 / 256, 256, 0, stream>>>(y, i1w, i1b, z);

    // conv2
    k_conv3x3<128><<<cgrid, 128, 0, stream>>>(z, w2f, y, 0);

    // irnn2
    k_irnn4<<<262144 / 256, 256, 0, stream>>>(y, i2w, i2b, z);

    // conv3 + relu
    k_conv3x3<128><<<cgrid, 128, 0, stream>>>(z, w3f, y, 1);

    // 1x1 conv + sigmoid
    long npix = (long)BB * HH * WWD;
    k_out<<<(unsigned)((npix + 255) / 256), 256, 0, stream>>>(y, w_o, mask);
}
